// srm_65609920414506
// MI455X (gfx1250) — matrix operand order repaired and checked by reading, not compiled, not run on hardware
//
#include <hip/hip_runtime.h>
#include <hip/hip_bf16.h>

// ---------------------------------------------------------------------------
// Types for CDNA5 WMMA (wave32, 16x16x32 bf16 -> f32)
// ---------------------------------------------------------------------------
typedef __attribute__((ext_vector_type(16))) __bf16 v16bf;
typedef __attribute__((ext_vector_type(8)))  __bf16 v8bf;
typedef __attribute__((ext_vector_type(8)))  float  v8f;

static __device__ inline v8f wmma_bf16(v16bf a, v16bf b, v8f c) {
    return __builtin_amdgcn_wmma_f32_16x16x32_bf16(
        /*neg_a=*/false, a, /*neg_b=*/false, b,
        /*c_mod=*/(short)0, c, /*reuse_a=*/false, /*reuse_b=*/false);
}

static __device__ inline v16bf cat16(v8bf lo, v8bf hi) {
    return __builtin_shufflevector(lo, hi, 0,1,2,3,4,5,6,7,8,9,10,11,12,13,14,15);
}
static __device__ inline v8f zerof8() {
    v8f z;
#pragma unroll
    for (int i = 0; i < 8; ++i) z[i] = 0.0f;
    return z;
}

// ---------------------------------------------------------------------------
// Problem constants (reference: BA=1, CH=128, NUM=12, S=192)
// ---------------------------------------------------------------------------
#define NUMP 12
#define SS   192
#define CHN  128

// LDS layout for the fused per-patch kernel (~294 KB of the 320 KB WGP
// budget, dynamic shared). Strides padded by 8 bf16 (16 B) to break bank
// conflicts while keeping every fragment load 16-byte aligned (ds_load_b128).
// Input tile has a 257th "zero row" (ZSP) and the pooled tile a 65th zero
// slot (ZPS): out-of-bounds conv halo taps redirect their LDS address there,
// so padding costs one v_cndmask on the address instead of 32 VALU selects.
#define W1_STRIDE 2312                 // bf16 per oc row (2304 + 8 pad)
#define IN_STRIDE 264                  // bf16 per spatial row (256 + 8 pad)
#define W2_STRIDE 296                  // bf16 per oc row (288 + 8 pad)
#define ZSP       256                  // zero spatial slot (conv1 A-frags)
#define ZPS       64                   // zero pooled slot (conv2 A-frags)
#define W1_OFF    0
#define IN_OFF    (32 * W1_STRIDE * 2)            // 147968
#define ACT1_OFF  IN_OFF                           // f32 [256][32], aliases input
#define POOL_OFF  (IN_OFF + 256 * 32 * 4)          // bf16 [65][32], aliases input
#define W2_OFF    (IN_OFF + 257 * IN_STRIDE * 2)   // 283664
#define RED_OFF   (W2_OFF + 16 * W2_STRIDE * 2)    // 293136
#define K1_LDS    (RED_OFF + 256 * 4)              // 294160

// ---------------------------------------------------------------------------
// Kernel 0: fold FC1+FC2 (no nonlinearity between them) into one 1024-vector,
// and zero the loss accumulators.   weff = lw2 @ lw1 ; beff = lw2.lb1 + lb2
// ---------------------------------------------------------------------------
extern "C" __global__ __launch_bounds__(256)
void fold_fc_kernel(const float* __restrict__ lw1, const float* __restrict__ lb1,
                    const float* __restrict__ lw2, const float* __restrict__ lb2,
                    float* __restrict__ weff, float* __restrict__ beff,
                    float* __restrict__ accums)
{
    int k = blockIdx.x * 256 + threadIdx.x;     // 0..1023
    float a = 0.0f;
    for (int j = 0; j < 256; ++j) a = fmaf(lw2[j], lw1[j * 1024 + k], a);
    weff[k] = a;
    if (k == 0) {
        float bb = lb2[0];
        for (int j = 0; j < 256; ++j) bb = fmaf(lw2[j], lb1[j], bb);
        beff[0]   = bb;
        accums[0] = 0.0f;
        accums[1] = 0.0f;
    }
}

// ---------------------------------------------------------------------------
// Kernel 1: fused per-patch VGG scorer. One workgroup (8 wave32) per patch.
//   conv1: implicit GEMM, M=256 spatial, N=32 oc, K=2304 (9 taps x 256 ic),
//          bf16 WMMA 16x16x32, A/B fragments built with ds_load_b128 pairs.
//   pool 2x2 -> conv2: M=64, N=16, K=288 WMMA GEMM.
//   folded FC: 1024-dot with weff + sigmoid.
// ---------------------------------------------------------------------------
extern "C" __global__ __launch_bounds__(256)
void vgg_score_kernel(const float* __restrict__ xf, const float* __restrict__ xr,
                      const float* __restrict__ w1, const float* __restrict__ b1,
                      const float* __restrict__ g1, const float* __restrict__ be1,
                      const float* __restrict__ m1, const float* __restrict__ v1,
                      const float* __restrict__ w2, const float* __restrict__ b2,
                      const float* __restrict__ g2, const float* __restrict__ be2,
                      const float* __restrict__ m2, const float* __restrict__ v2,
                      const float* __restrict__ weff, const float* __restrict__ beff,
                      float* __restrict__ out)
{
    extern __shared__ __align__(16) char smem[];
    __bf16* w1l  = (__bf16*)(smem + W1_OFF);
    __bf16* inl  = (__bf16*)(smem + IN_OFF);
    float*  act1 = (float*) (smem + ACT1_OFF);
    __bf16* pooll= (__bf16*)(smem + POOL_OFF);
    __bf16* w2l  = (__bf16*)(smem + W2_OFF);
    float*  red  = (float*) (smem + RED_OFF);

    const int tid  = threadIdx.x;
    const int lane = tid & 31;
    const int wv   = tid >> 5;          // wave 0..7
    const int colx = lane & 15;         // N column / x coordinate
    const int hf   = lane >> 4;         // lane half (0/1)

    const int p   = blockIdx.x;                 // patch: v*144 + py*12 + px
    const int vI  = p / (NUMP * NUMP);
    const int py  = (p / NUMP) % NUMP;
    const int px  = p % NUMP;
    const int gy0 = py * 16, gx0 = px * 16;

    // ---- stage w1 -> LDS bf16 [oc][(ky*3+kx)*256 + ic] (coalesced reads) ----
    for (int i = tid; i < 32 * 2304; i += 256) {
        int oc = i / 2304, r = i % 2304;
        int ic = r / 9,    kp = r % 9;
        w1l[oc * W1_STRIDE + kp * 256 + ic] = (__bf16)w1[i];
    }
    // ---- stage w2 -> LDS bf16 [oc][(ky*3+kx)*32 + ic] ----
    for (int i = tid; i < 16 * 288; i += 256) {
        int oc = i / 288, r = i % 288;
        int ic = r / 9,   kp = r % 9;
        w2l[oc * W2_STRIDE + kp * 32 + ic] = (__bf16)w2[i];
    }
    // ---- zero row for halo taps ----
    if (tid < 128) {
        ((short*)(inl + ZSP * IN_STRIDE))[tid * 2]     = 0;
        ((short*)(inl + ZSP * IN_STRIDE))[tid * 2 + 1] = 0;
    }
    // ---- stage the 256ch x 16x16 patch -> LDS bf16 [y*16+x][ic] ----
    for (int i = tid; i < 256 * 256; i += 256) {
        int c  = i >> 8;           // channel 0..255
        int sp = i & 255;          // spatial y*16+x (consecutive tid -> coalesced x)
        int y = sp >> 4, x = sp & 15;
        float val;
        if (c < CHN) val = xf[(((size_t)vI * CHN + c) * SS + (gy0 + y)) * SS + (gx0 + x)];
        else         val = xr[(((size_t)(c - CHN)) * SS + (gy0 + y)) * SS + (gx0 + x)];
        inl[sp * IN_STRIDE + c] = (__bf16)val;
    }
    __syncthreads();

    // ---- conv1 implicit GEMM: wave wv owns M-tiles {2wv, 2wv+1} x N-tiles {0,1}
    // M-tile m covers spatial row y=m (M = y*16 + x). K = kp*256 + ic.
    const int mt0 = 2 * wv, mt1 = 2 * wv + 1;
    v8f acc00 = zerof8(), acc01 = zerof8(), acc10 = zerof8(), acc11 = zerof8();

    float sc1[2], sh1[2];
#pragma unroll
    for (int n = 0; n < 2; ++n) {
        int oc = n * 16 + colx;
        float s = g1[oc] * rsqrtf(v1[oc] + 1e-5f);
        sc1[n] = s;
        sh1[n] = (b1[oc] - m1[oc]) * s + be1[oc];
    }

    for (int kp = 0; kp < 9; ++kp) {
        const int ky = kp / 3 - 1, kx = kp % 3 - 1;
        const int ix  = colx + kx;
        const bool xok  = (unsigned)ix < 16u;
        const int iy0 = mt0 + ky;  const bool ok0 = xok && ((unsigned)iy0 < 16u);
        const int iy1 = mt1 + ky;  const bool ok1 = xok && ((unsigned)iy1 < 16u);
        const int sp0 = ok0 ? (iy0 * 16 + ix) : ZSP;    // halo -> zero row
        const int sp1 = ok1 ? (iy1 * 16 + ix) : ZSP;
        const __bf16* arow0 = inl + sp0 * IN_STRIDE + hf * 8;
        const __bf16* arow1 = inl + sp1 * IN_STRIDE + hf * 8;
#pragma unroll
        for (int icb = 0; icb < 8; ++icb) {
            const int ic0 = icb * 32;                   // A frag: K(l,j) layout
            // A fragments: two ds_load_b128 each (8 consecutive ic, then +16 ic)
            v16bf a0 = cat16(*(const v8bf*)(arow0 + ic0),
                             *(const v8bf*)(arow0 + ic0 + 16));
            v16bf a1 = cat16(*(const v8bf*)(arow1 + ic0),
                             *(const v8bf*)(arow1 + ic0 + 16));
            // B fragments: 8 consecutive K, then +16 K, at fixed oc (matches A)
            const int kb = kp * 256 + ic0 + hf * 8;
            const __bf16* bp0 = w1l + colx * W1_STRIDE + kb;
            const __bf16* bp1 = w1l + (16 + colx) * W1_STRIDE + kb;
            v16bf bf0 = cat16(*(const v8bf*)bp0, *(const v8bf*)(bp0 + 16));
            v16bf bf1 = cat16(*(const v8bf*)bp1, *(const v8bf*)(bp1 + 16));

            acc00 = wmma_bf16(a0, bf0, acc00);
            acc01 = wmma_bf16(a0, bf1, acc01);
            acc10 = wmma_bf16(a1, bf0, acc10);
            acc11 = wmma_bf16(a1, bf1, acc11);
        }
    }
    __syncthreads();   // everyone done reading inl; act1 aliases it

    // ---- bias + BN + ReLU, scatter C fragments to act1[spatial][oc] (f32) ----
#pragma unroll
    for (int j = 0; j < 8; ++j) {
        const int sp0 = mt0 * 16 + hf * 8 + j;   // C layout: M = j + 8*(lane/16)
        const int sp1 = mt1 * 16 + hf * 8 + j;
        act1[sp0 * 32 + colx]      = fmaxf(acc00[j] * sc1[0] + sh1[0], 0.0f);
        act1[sp0 * 32 + 16 + colx] = fmaxf(acc01[j] * sc1[1] + sh1[1], 0.0f);
        act1[sp1 * 32 + colx]      = fmaxf(acc10[j] * sc1[0] + sh1[0], 0.0f);
        act1[sp1 * 32 + 16 + colx] = fmaxf(acc11[j] * sc1[1] + sh1[1], 0.0f);
    }
    __syncthreads();

    // ---- 2x2 average pool -> pooll[ps=y2*8+x2][oc] (bf16), + zero slot ----
    for (int i = tid; i < 64 * 32; i += 256) {
        int ps = i >> 5, oc = i & 31;
        int y2 = ps >> 3, x2 = ps & 7;
        int s00 = (2 * y2 * 16 + 2 * x2) * 32 + oc;
        float m = 0.25f * (act1[s00] + act1[s00 + 32] +
                           act1[s00 + 16 * 32] + act1[s00 + 16 * 32 + 32]);
        pooll[ps * 32 + oc] = (__bf16)m;
    }
    if (tid < 32) pooll[ZPS * 32 + tid] = (__bf16)0.0f;
    __syncthreads();

    // ---- conv2 WMMA GEMM (waves 0..3, one 16-row M-tile each) + folded FC ----
    float partial = 0.0f;
    if (wv < 4) {
        v8f acc2 = zerof8();
        const float s2c = g2[colx] * rsqrtf(v2[colx] + 1e-5f);
        const float s2h = (b2[colx] - m2[colx]) * s2c + be2[colx];
        const int ps = wv * 16 + colx;         // A row for this lane
        const int y2 = ps >> 3, x2 = ps & 7;
#pragma unroll
        for (int kp = 0; kp < 9; ++kp) {
            const int ky = kp / 3 - 1, kx = kp % 3 - 1;
            const int iy = y2 + ky, ixx = x2 + kx;
            const bool ok = ((unsigned)iy < 8u) && ((unsigned)ixx < 8u);
            const int sp = ok ? (iy * 8 + ixx) : ZPS;   // halo -> zero slot
            const __bf16* ap = pooll + sp * 32 + hf * 8;
            v16bf a2 = cat16(*(const v8bf*)ap, *(const v8bf*)(ap + 16));
            const __bf16* bp = w2l + colx * W2_STRIDE + kp * 32 + hf * 8;
            v16bf bf2 = cat16(*(const v8bf*)bp, *(const v8bf*)(bp + 16));
            acc2 = wmma_bf16(a2, bf2, acc2);
        }
#pragma unroll
        for (int j = 0; j < 8; ++j) {
            const int psj = wv * 16 + hf * 8 + j;       // flatten: oc*64 + spatial
            const float hval = fmaxf(acc2[j] * s2c + s2h, 0.0f);
            partial = fmaf(hval, weff[colx * 64 + psj], partial);
        }
    }

    // ---- workgroup reduction + sigmoid ----
    red[tid] = partial;
    __syncthreads();
    for (int s = 128; s > 0; s >>= 1) {
        if (tid < s) red[tid] += red[tid + s];
        __syncthreads();
    }
    if (tid == 0) {
        const float z = red[0] + beff[0];
        out[p] = 1.0f / (1.0f + __expf(-z));
    }
}

// ---------------------------------------------------------------------------
// Kernel 2: one memory-bound pass over xf per 16x16 tile (144 workgroups).
// With A = sum_k ck*feat_k and Sm = sum_k feat_k:
//   n1^2 = |A|^2/sck^2 ; n0^2 = |Sm-A|^2/(12-sck)^2 ; <t1,t0> = <A,Sm-A>/...
//   d1 = sck*n1^2 ; d0 = (12-sck)*n0^2 ; nf1^2 = sum_k ck^2 |feat_k|^2 ; etc.
// ---------------------------------------------------------------------------
extern "C" __global__ __launch_bounds__(256)
void stats_kernel(const float* __restrict__ xf, const float* __restrict__ scores,
                  float* __restrict__ accums)
{
    __shared__ float red[15 * 256];
    const int tid  = threadIdx.x;
    const int tile = blockIdx.x;           // ny*12 + nx
    const int ny = tile / NUMP, nx = tile % NUMP;

    float ck[12];
    float sck = 0.0f;
#pragma unroll
    for (int k = 0; k < 12; ++k) { ck[k] = scores[k * 144 + tile]; sck += ck[k]; }

    const size_t viewStride = (size_t)CHN * SS * SS;
    float a2 = 0.0f, s2 = 0.0f, as = 0.0f;
    float F[12];
#pragma unroll
    for (int k = 0; k < 12; ++k) F[k] = 0.0f;

    for (int i = tid; i < CHN * 16 * 16; i += 256) {
        const int c = i >> 8;
        const int r = i & 255;
        const int y = r >> 4, x = r & 15;
        const size_t off = ((size_t)c * SS + (ny * 16 + y)) * SS + (nx * 16 + x);
        float a = 0.0f, s = 0.0f;
#pragma unroll
        for (int k = 0; k < 12; ++k) {
            const float f = xf[k * viewStride + off];
            a = fmaf(ck[k], f, a);
            s += f;
            F[k] = fmaf(f, f, F[k]);
        }
        a2 = fmaf(a, a, a2);
        s2 = fmaf(s, s, s2);
        as = fmaf(a, s, as);
    }

    red[0 * 256 + tid] = a2;
    red[1 * 256 + tid] = s2;
    red[2 * 256 + tid] = as;
#pragma unroll
    for (int k = 0; k < 12; ++k) red[(3 + k) * 256 + tid] = F[k];
    __syncthreads();
    for (int s = 128; s > 0; s >>= 1) {
        if (tid < s) {
#pragma unroll
            for (int q = 0; q < 15; ++q) red[q * 256 + tid] += red[q * 256 + tid + s];
        }
        __syncthreads();
    }

    if (tid == 0) {
        const float A2 = red[0], S2 = red[256], AS = red[512];
        const float inv1 = 1.0f / sck;
        const float inv0 = 1.0f / (12.0f - sck);
        const float n1sq = A2 * inv1 * inv1;
        const float n0sq = (S2 - 2.0f * AS + A2) * inv0 * inv0;
        const float n1 = sqrtf(n1sq), n0 = sqrtf(n0sq);
        const float t1t0 = (AS - A2) * inv1 * inv0;
        const float c1 = t1t0 / fmaxf(n1 * n0, 1e-8f);
        const float d1 = sck * n1sq;
        const float d0 = (12.0f - sck) * n0sq;
        float nf1s = 0.0f, nf0s = 0.0f;
#pragma unroll
        for (int k = 0; k < 12; ++k) {
            const float Fk = red[(3 + k) * 256];
            nf1s = fmaf(ck[k] * ck[k], Fk, nf1s);
            nf0s = fmaf((1.0f - ck[k]) * (1.0f - ck[k]), Fk, nf0s);
        }
        const float nf1 = sqrtf(nf1s), nf0 = sqrtf(nf0s);
        const float rt12 = 3.4641016151377544f;
        const float c2 = d1 / fmaxf(rt12 * n1 * nf1, 1e-8f)
                       + d0 / fmaxf(rt12 * n0 * nf0, 1e-8f);
        atomicAdd(&accums[0], c1);
        atomicAdd(&accums[1], 2.0f - c2);
    }
}

extern "C" __global__ void finalize_kernel(const float* __restrict__ accums,
                                           float* __restrict__ out)
{
    out[1728] = 1.0f - accums[0] * (1.0f / 144.0f);
    out[1729] = 1.0f - accums[1] * (1.0f / 144.0f);
}

// ---------------------------------------------------------------------------
// Host launcher
// ---------------------------------------------------------------------------
extern "C" void kernel_launch(void* const* d_in, const int* in_sizes, int n_in,
                              void* d_out, int out_size, void* d_ws, size_t ws_size,
                              hipStream_t stream)
{
    const float* xf  = (const float*)d_in[0];
    const float* xr  = (const float*)d_in[1];
    const float* w1  = (const float*)d_in[2];
    const float* b1  = (const float*)d_in[3];
    const float* g1  = (const float*)d_in[4];
    const float* be1 = (const float*)d_in[5];
    const float* m1  = (const float*)d_in[6];
    const float* v1  = (const float*)d_in[7];
    const float* w2  = (const float*)d_in[8];
    const float* b2  = (const float*)d_in[9];
    const float* g2  = (const float*)d_in[10];
    const float* be2 = (const float*)d_in[11];
    const float* m2  = (const float*)d_in[12];
    const float* v2  = (const float*)d_in[13];
    const float* lw1 = (const float*)d_in[14];
    const float* lb1 = (const float*)d_in[15];
    const float* lw2 = (const float*)d_in[16];
    const float* lb2 = (const float*)d_in[17];
    float* out = (float*)d_out;

    float* weff   = (float*)d_ws;      // [1024]
    float* beff   = weff + 1024;       // [1]
    float* accums = weff + 1025;       // [2] (c1 sum, (2-c2) sum)

    // Allow >64KB dynamic LDS for the fused kernel (gfx1250: 320KB/WGP).
    (void)hipFuncSetAttribute((const void*)vgg_score_kernel,
                              hipFuncAttributeMaxDynamicSharedMemorySize, K1_LDS);

    fold_fc_kernel<<<4, 256, 0, stream>>>(lw1, lb1, lw2, lb2, weff, beff, accums);
    vgg_score_kernel<<<12 * NUMP * NUMP, 256, K1_LDS, stream>>>(
        xf, xr, w1, b1, g1, be1, m1, v1, w2, b2, g2, be2, m2, v2, weff, beff, out);
    stats_kernel<<<NUMP * NUMP, 256, 0, stream>>>(xf, out, accums);
    finalize_kernel<<<1, 1, 0, stream>>>(accums, out);
}
